// DescriptorLoss_35167192219894
// MI455X (gfx1250) — matrix operand order repaired and checked by reading, not compiled, not run on hardware
//
#include <hip/hip_runtime.h>

// Problem constants (match reference)
#define B_    2
#define D_    128       // K dimension
#define P_    4096      // H*W pixels per image
#define TILE  128       // workgroup C tile: 128x128
#define KPAD  136       // LDS row stride in bf16 elems (16B-aligned rows, bank-skewed)
#define NWAVES 8

#define MARGIN_POS 1.0f
#define MARGIN_NEG 0.2f
#define LAMBDA_D   250.0f
// mean divisor = B*P*P = 2*4096*4096 = 2^25 (exact in f32)
#define INV_COUNT  (1.0f / 33554432.0f)

typedef __attribute__((ext_vector_type(16))) __bf16 v16bf;
typedef __attribute__((ext_vector_type(8)))  float  v8f;
typedef __attribute__((ext_vector_type(4)))  int    v4i;
typedef __attribute__((ext_vector_type(4)))  float  v4f;

union Frag {
    v4i   i4[2];   // 32 bytes = two ds_load_b128
    v16bf bf;      // 16 bf16 operand for v_wmma_f32_16x16x32_bf16
};

__device__ __forceinline__ unsigned short f32_to_bf16_rne(float f) {
    unsigned int u = __float_as_uint(f);
    unsigned int r = u + 0x7FFFu + ((u >> 16) & 1u);   // round-to-nearest-even
    return (unsigned short)(r >> 16);
}
__device__ __forceinline__ float bf16_to_f32(unsigned short h) {
    return __uint_as_float(((unsigned int)h) << 16);
}

__global__ void zero_out_kernel(float* out) {
    if (threadIdx.x == 0 && blockIdx.x == 0) out[0] = 0.0f;
}

__global__ __launch_bounds__(256)
void desc_loss_wmma_kernel(const float* __restrict__ d0,
                           const float* __restrict__ d1,
                           const unsigned char* __restrict__ mask,
                           float* __restrict__ out)
{
    // bf16 hi/lo splits of A (desc0 tile) and B (desc1 tile), stored [pixel][k]
    __shared__ unsigned short sAh[TILE * KPAD];
    __shared__ unsigned short sAl[TILE * KPAD];
    __shared__ unsigned short sBh[TILE * KPAD];
    __shared__ unsigned short sBl[TILE * KPAD];
    __shared__ float sPartial[NWAVES];

    const int tid  = threadIdx.x;
    const int lane = tid & 31;
    const int wave = tid >> 5;
    const int nBase = blockIdx.x * TILE;   // pixel index into image 1
    const int mBase = blockIdx.y * TILE;   // pixel index into image 0
    const int b     = blockIdx.z;

    const float* a_g = d0 + (size_t)b * D_ * P_ + mBase;  // [k][m] rows of 4096
    const float* b_g = d1 + (size_t)b * D_ * P_ + nBase;  // [k][n]
    const unsigned char* msk = mask + (size_t)b * P_ * P_;

    // Prefetch this block's mask tile (128x128 bytes) into cache while we GEMM.
    if (tid < TILE) {
        const unsigned char* mrow = msk + (size_t)(mBase + tid) * P_ + nBase;
        __builtin_prefetch(mrow, 0, 1);        // global_prefetch_b8
        __builtin_prefetch(mrow + 64, 0, 1);
    }

    // ---- Stage descriptors: f32 -> (bf16 hi, bf16 lo), transposed into LDS ----
    // Each iteration: one float4 of A and one of B per thread (coalesced b128 loads).
    for (int idx = tid; idx < (TILE * D_) / 4; idx += 256) {
        const int k  = idx / (TILE / 4);
        const int m4 = (idx % (TILE / 4)) * 4;
        v4f av = *(const v4f*)(a_g + (size_t)k * P_ + m4);
        v4f bv = *(const v4f*)(b_g + (size_t)k * P_ + m4);
#pragma unroll
        for (int j = 0; j < 4; ++j) {
            float a = av[j];
            unsigned short ah = f32_to_bf16_rne(a);
            unsigned short al = f32_to_bf16_rne(a - bf16_to_f32(ah));
            sAh[(m4 + j) * KPAD + k] = ah;
            sAl[(m4 + j) * KPAD + k] = al;
            float bb = bv[j];
            unsigned short bh = f32_to_bf16_rne(bb);
            unsigned short bl = f32_to_bf16_rne(bb - bf16_to_f32(bh));
            sBh[(m4 + j) * KPAD + k] = bh;
            sBl[(m4 + j) * KPAD + k] = bl;
        }
    }
    __syncthreads();

    // ---- GEMM: each wave computes a 16x128 strip = 8 C tiles of 16x16 ----
    const v8f vzero = {0.f, 0.f, 0.f, 0.f, 0.f, 0.f, 0.f, 0.f};
    v8f acc[8];
#pragma unroll
    for (int t = 0; t < 8; ++t) acc[t] = vzero;

    const int mSub = wave * 16;
    const int lrow = mSub + (lane & 15);          // A row this lane supplies
    const int aOff = (lane < 16) ? 0 : 8;         // A: K 0-7/16-23 vs 8-15/24-31
    const int bOff = (lane < 16) ? 0 : 8;         // B: K 0-7/16-23 vs 8-15/24-31

#pragma unroll
    for (int kc = 0; kc < D_; kc += 32) {
        Frag ah, al;
        {
            const unsigned short* pa  = &sAh[lrow * KPAD + kc + aOff];
            const unsigned short* pal = &sAl[lrow * KPAD + kc + aOff];
            ah.i4[0] = *(const v4i*)(pa);         // K = aOff .. aOff+7
            ah.i4[1] = *(const v4i*)(pa + 16);    // K = aOff+16 .. aOff+23
            al.i4[0] = *(const v4i*)(pal);
            al.i4[1] = *(const v4i*)(pal + 16);
        }
#pragma unroll
        for (int nt = 0; nt < 8; ++nt) {
            const int brow = nt * 16 + (lane & 15);  // B column this lane supplies
            Frag bh, bl;
            const unsigned short* pb  = &sBh[brow * KPAD + kc + bOff];
            const unsigned short* pbl = &sBl[brow * KPAD + kc + bOff];
            bh.i4[0] = *(const v4i*)(pb);          // K = bOff .. bOff+7
            bh.i4[1] = *(const v4i*)(pb + 16);     // K = bOff+16 .. bOff+23
            bl.i4[0] = *(const v4i*)(pbl);
            bl.i4[1] = *(const v4i*)(pbl + 16);
            // bf16x3: Ah*Bh + Ah*Bl + Al*Bh  (near-fp32 accuracy)
            acc[nt] = __builtin_amdgcn_wmma_f32_16x16x32_bf16(
                false, ah.bf, false, bh.bf, (short)0, acc[nt], false, false);
            acc[nt] = __builtin_amdgcn_wmma_f32_16x16x32_bf16(
                false, ah.bf, false, bl.bf, (short)0, acc[nt], false, false);
            acc[nt] = __builtin_amdgcn_wmma_f32_16x16x32_bf16(
                false, al.bf, false, bh.bf, (short)0, acc[nt], false, false);
        }
    }

    // ---- Epilogue: masked hinge loss + reduction ----
    // C/D layout: VGPR v -> M = mSub + v + (lane>=16 ? 8 : 0), N = nt*16 + (lane&15)
    const int nLane = lane & 15;
    const int mAdd  = (lane < 16) ? 0 : 8;
    float lsum = 0.0f;
#pragma unroll
    for (int nt = 0; nt < 8; ++nt) {
        const int gn = nBase + nt * 16 + nLane;
#pragma unroll
        for (int v = 0; v < 8; ++v) {
            const int gm = mBase + mSub + mAdd + v;
            const float dot = acc[nt][v];
            const float pos = fmaxf(0.0f, MARGIN_POS - dot) * LAMBDA_D;
            const float neg = fmaxf(0.0f, dot - MARGIN_NEG);
            const bool mv = msk[(size_t)gm * P_ + gn] != 0;
            lsum += mv ? pos : neg;
        }
    }

    // wave32 tree reduction, then cross-wave via LDS, then one global atomic
#pragma unroll
    for (int off = 16; off > 0; off >>= 1)
        lsum += __shfl_down(lsum, off, 32);
    if (lane == 0) sPartial[wave] = lsum;
    __syncthreads();
    if (tid == 0) {
        float s = 0.0f;
#pragma unroll
        for (int w = 0; w < NWAVES; ++w) s += sPartial[w];
        atomicAdd(out, s * INV_COUNT);
    }
}

extern "C" void kernel_launch(void* const* d_in, const int* in_sizes, int n_in,
                              void* d_out, int out_size, void* d_ws, size_t ws_size,
                              hipStream_t stream) {
    (void)in_sizes; (void)n_in; (void)out_size; (void)d_ws; (void)ws_size;
    const float* d0 = (const float*)d_in[0];
    const float* d1 = (const float*)d_in[1];
    const unsigned char* mask = (const unsigned char*)d_in[2];  // jax bool_ = 1 byte
    float* out = (float*)d_out;

    zero_out_kernel<<<1, 32, 0, stream>>>(out);

    dim3 grid(P_ / TILE, P_ / TILE, B_);   // 32 x 32 x 2 = 2048 workgroups
    desc_loss_wmma_kernel<<<grid, 256, 0, stream>>>(d0, d1, mask, out);
}
